// Model_40243843563714
// MI455X (gfx1250) — matrix operand order repaired and checked by reading, not compiled, not run on hardware
//
#include <hip/hip_runtime.h>

// ============================================================================
// GRU forward on MI455X (gfx1250): WMMA f16 (f32 accum) everywhere.
//   phase0: convert weights f32 -> f16 into workspace.
//   phase1 (if ws big enough): time-parallel ig = (x@W_in^T + b_in)@W_ih^T + b_ih
//           stored in WMMA C-layout tiles (f16) -> 805 MB workspace.
//   phase2: 16 persistent workgroups, each owns 16 batch rows; T=2048 scan with
//           double-buffered h in LDS (f16), W_hh fragments pinned in VGPRs,
//           24 WMMA/step/wave, ig tiles prefetched one step ahead,
//           1 barrier/step, h_old carried in registers.
//   output: h_final @ W_out^T + b_out fused into phase2 tail.
// ============================================================================

namespace {

constexpr int kB   = 256;
constexpr int kT   = 2048;
constexpr int kIN  = 64;
constexpr int kH   = 256;
constexpr int kG3  = 768;        // 3*H
constexpr int kOUT = 32;

constexpr int BT      = 16;              // batch-tile rows per workgroup (WMMA M)
constexpr int NWG     = kB / BT;         // 16 workgroups
constexpr int NTILES  = kG3 / 16;        // 48 gate column tiles
constexpr int WAVES   = 16;
constexpr int THREADS = WAVES * 32;      // 512

typedef __attribute__((ext_vector_type(16))) _Float16 v16h;
typedef __attribute__((ext_vector_type(8)))  _Float16 h8;
typedef __attribute__((ext_vector_type(8)))  float    v8f;

// ---- workspace layout (bytes) ----
constexpr size_t WS_WIN   = 0;
constexpr size_t WS_WIH   = WS_WIN  + (size_t)kH  * kIN * 2;
constexpr size_t WS_WHH   = WS_WIH  + (size_t)kG3 * kH  * 2;
constexpr size_t WS_WOUT  = WS_WHH  + (size_t)kG3 * kH  * 2;
constexpr size_t WS_WEND  = WS_WOUT + (size_t)kOUT * kH * 2;   // ~836 KB
constexpr size_t WS_IG    = (size_t)1 << 20;                   // 1 MB aligned
constexpr size_t IG_BYTES = (size_t)kT * NWG * NTILES * 256 * 2; // 805,306,368

__device__ __forceinline__ v8f vbroadcast(float s) {
  v8f r;
#pragma unroll
  for (int i = 0; i < 8; ++i) r[i] = s;
  return r;
}

// A-operand fragment (16x32 f16, M = lane%16 handled by caller via rowPtr):
// per ISA: elems 0..7  = K(kOff + 8*(lane/16) .. +7)
//          elems 8..15 = K(kOff + 16 + 8*(lane/16) .. +7)
__device__ __forceinline__ v16h load_frag_a(const _Float16* rowPtr, int kOff) {
  int h8off = ((threadIdx.x & 31) >> 4) * 8;
  h8 a = *(const h8*)(rowPtr + kOff + h8off);
  h8 b = *(const h8*)(rowPtr + kOff + h8off + 16);
  return __builtin_shufflevector(a, b, 0, 1, 2, 3, 4, 5, 6, 7,
                                       8, 9, 10, 11, 12, 13, 14, 15);
}

// B-operand fragment (32x16 f16, N = lane%16 -> row of W since B[k][n] = W[n][k]):
// per lane: elems 0..7 = K(kOff + 8*(lane/16) .. +7), elems 8..15 = K(kOff + 16 + 8*(lane/16) .. +7).
__device__ __forceinline__ v16h load_frag_b(const _Float16* rowPtr, int kOff) {
  int c0 = kOff + ((threadIdx.x & 31) >> 4) * 8;
  h8 a = *(const h8*)(rowPtr + c0);
  h8 b = *(const h8*)(rowPtr + c0 + 16);
  return __builtin_shufflevector(a, b, 0, 1, 2, 3, 4, 5, 6, 7,
                                       8, 9, 10, 11, 12, 13, 14, 15);
}

__device__ __forceinline__ v8f wmma16(v16h a, v16h b, v8f c) {
  return __builtin_amdgcn_wmma_f32_16x16x32_f16(false, a, false, b,
                                                (short)0, c, false, false);
}

// fast sigmoid / tanh built from v_exp_f32 + v_rcp_f32 (TRANS ops co-execute
// with WMMA); tanh input clamped so exp cannot overflow to Inf -> NaN.
__device__ __forceinline__ float sigm(float v) {
  float e = __expf(-v);
  return __builtin_amdgcn_rcpf(1.0f + e);
}
__device__ __forceinline__ float tanh_f(float a) {
  float ac = fminf(fmaxf(a, -15.0f), 15.0f);
  float e  = __expf(-2.0f * ac);
  return (1.0f - e) * __builtin_amdgcn_rcpf(1.0f + e);
}

// ---------------------------------------------------------------------------
// phase0: f32 -> f16 weight conversion
// ---------------------------------------------------------------------------
__global__ void cvt_weights_kernel(const float* __restrict__ Win,
                                   const float* __restrict__ Wih,
                                   const float* __restrict__ Whh,
                                   const float* __restrict__ Wout,
                                   _Float16* __restrict__ dWin,
                                   _Float16* __restrict__ dWih,
                                   _Float16* __restrict__ dWhh,
                                   _Float16* __restrict__ dWout) {
  int i = blockIdx.x * 256 + threadIdx.x;
  if (i < kH * kIN)  dWin[i]  = (_Float16)Win[i];
  if (i < kG3 * kH)  { dWih[i] = (_Float16)Wih[i]; dWhh[i] = (_Float16)Whh[i]; }
  if (i < kOUT * kH) dWout[i] = (_Float16)Wout[i];
}

// ---------------------------------------------------------------------------
// phase1: ig[t][wg][48 tiles] in C-layout lane order (h8 per lane), f16.
// grid = kT * NWG blocks x 512 threads. Wave w -> u-tile w, ig-tiles {w,16+w,32+w}.
// ---------------------------------------------------------------------------
__global__ __launch_bounds__(THREADS, 1)
void ig_kernel(const float* __restrict__ x,
               const float* __restrict__ b_in,
               const float* __restrict__ b_ih,
               const _Float16* __restrict__ wWin,
               const _Float16* __restrict__ wWih,
               h8* __restrict__ igws) {
  __shared__ __align__(16) _Float16 xs[BT * kIN];
  __shared__ __align__(16) _Float16 us[BT * kH];

  const int wg   = blockIdx.x & (NWG - 1);
  const int t    = blockIdx.x >> 4;
  const int tid  = threadIdx.x;
  const int lane = tid & 31;
  const int w    = tid >> 5;
  const int nIdx = lane & 15;
  const int mRow = lane & 15;
  const int mBase = (lane >> 4) << 3;

  // cooperative x tile load: 16 rows x 64 cols, f32 -> f16
  {
    int r = tid >> 5;
    int c = (tid & 31) * 2;
    float2 v = *(const float2*)(x + ((size_t)(wg * BT + r) * kT + t) * kIN + c);
    xs[r * kIN + c]     = (_Float16)v.x;
    xs[r * kIN + c + 1] = (_Float16)v.y;
  }
  __syncthreads();

  // u = x @ W_in^T + b_in  (wave w -> hidden cols w*16..w*16+15)
  {
    v8f acc = vbroadcast(b_in[w * 16 + nIdx]);
#pragma unroll
    for (int k = 0; k < 2; ++k) {
      v16h A = load_frag_a(xs + mRow * kIN, k * 32);
      v16h Bf = load_frag_b(wWin + (size_t)(w * 16 + nIdx) * kIN, k * 32);
      acc = wmma16(A, Bf, acc);
    }
#pragma unroll
    for (int v = 0; v < 8; ++v)
      us[(v + mBase) * kH + w * 16 + nIdx] = (_Float16)acc[v];
  }
  __syncthreads();

  // ig = u @ W_ih^T + b_ih ; 3 gate tiles per wave, shared A per K-step
  {
    v8f aR = vbroadcast(b_ih[w * 16 + nIdx]);
    v8f aZ = vbroadcast(b_ih[256 + w * 16 + nIdx]);
    v8f aN = vbroadcast(b_ih[512 + w * 16 + nIdx]);
    const _Float16* rR = wWih + (size_t)((0 * 16 + w) * 16 + nIdx) * kH;
    const _Float16* rZ = wWih + (size_t)((16 + w) * 16 + nIdx) * kH;
    const _Float16* rN = wWih + (size_t)((32 + w) * 16 + nIdx) * kH;
#pragma unroll
    for (int k = 0; k < 8; ++k) {
      v16h A = load_frag_a(us + mRow * kH, k * 32);
      aR = wmma16(A, load_frag_b(rR, k * 32), aR);
      aZ = wmma16(A, load_frag_b(rZ, k * 32), aZ);
      aN = wmma16(A, load_frag_b(rN, k * 32), aN);
    }
    size_t tb = ((size_t)t * NWG + wg) * NTILES;
    h8 oR, oZ, oN;
#pragma unroll
    for (int v = 0; v < 8; ++v) {
      oR[v] = (_Float16)aR[v]; oZ[v] = (_Float16)aZ[v]; oN[v] = (_Float16)aN[v];
    }
    igws[(tb + w) * 32 + lane]      = oR;
    igws[(tb + 16 + w) * 32 + lane] = oZ;
    igws[(tb + 32 + w) * 32 + lane] = oN;
  }
}

// ---------------------------------------------------------------------------
// phase2: persistent recurrence. grid = 16 blocks x 512 threads.
// USE_IG: read precomputed gate tiles (prefetched 1 step ahead);
// else compute u/ig inline each step.
// ---------------------------------------------------------------------------
template <bool USE_IG>
__global__ __launch_bounds__(THREADS, 1)
void rec_kernel(const float* __restrict__ x,
                const float* __restrict__ b_in,
                const float* __restrict__ b_ih,
                const float* __restrict__ b_n,
                const float* __restrict__ b_out,
                const _Float16* __restrict__ wWin,
                const _Float16* __restrict__ wWih,
                const _Float16* __restrict__ wWhh,
                const _Float16* __restrict__ wWout,
                const h8* __restrict__ igws,
                float* __restrict__ out) {
  // double-buffered hidden state: step t reads hs+(t&1)*BT*kH, writes the other
  __shared__ __align__(16) _Float16 hs[2 * BT * kH];
  __shared__ __align__(16) _Float16 xs[BT * kIN];
  __shared__ __align__(16) _Float16 us[BT * kH];

  const int wg   = blockIdx.x;
  const int tid  = threadIdx.x;
  const int lane = tid & 31;
  const int w    = tid >> 5;
  const int nIdx = lane & 15;
  const int mRow = lane & 15;
  const int mBase = (lane >> 4) << 3;

  for (int i = tid; i < BT * kH; i += THREADS) hs[i] = (_Float16)0.0f;

  const float bn  = b_n[w * 16 + nIdx];
  float biU = 0.f, biR = 0.f, biZ = 0.f, biN = 0.f;
  if (!USE_IG) {
    biU = b_in[w * 16 + nIdx];
    biR = b_ih[w * 16 + nIdx];
    biZ = b_ih[256 + w * 16 + nIdx];
    biN = b_ih[512 + w * 16 + nIdx];
  }

  // Pin W_hh fragments for this wave's 3 gate tiles across the whole scan:
  // 24 x v16h = 192 VGPRs of weights living in registers.
  v16h Bhh[3][8];
#pragma unroll
  for (int g = 0; g < 3; ++g) {
    const _Float16* row = wWhh + (size_t)((g * 16 + w) * 16 + nIdx) * kH;
#pragma unroll
    for (int k = 0; k < 8; ++k) Bhh[g][k] = load_frag_b(row, k * 32);
  }

  // h_old for this wave's columns carried in registers across steps
  float hold[8];
#pragma unroll
  for (int v = 0; v < 8; ++v) hold[v] = 0.0f;

  // prefetch ig tiles for step 0
  h8 r8, z8, n8;
  if (USE_IG) {
    size_t tb = (size_t)wg * NTILES;
    r8 = igws[(tb + w) * 32 + lane];
    z8 = igws[(tb + 16 + w) * 32 + lane];
    n8 = igws[(tb + 32 + w) * 32 + lane];
  }
  __syncthreads();

  for (int t = 0; t < kT; ++t) {
    const _Float16* hcur = hs + (t & 1) * (BT * kH);
    _Float16*       hnxt = hs + ((t + 1) & 1) * (BT * kH);

    v8f accR = vbroadcast(0.f), accZ = vbroadcast(0.f), accN = vbroadcast(0.f);
    h8 r8c, z8c, n8c;
    float igr[8], igz[8], ign[8];

    if (USE_IG) {
      r8c = r8; z8c = z8; n8c = n8;
      // issue next step's loads now; they complete under this step's WMMAs
      if (t + 1 < kT) {
        size_t tb = ((size_t)(t + 1) * NWG + wg) * NTILES;
        r8 = igws[(tb + w) * 32 + lane];
        z8 = igws[(tb + 16 + w) * 32 + lane];
        n8 = igws[(tb + 32 + w) * 32 + lane];
      }
    } else {
      // x tile -> LDS
      {
        int r = tid >> 5;
        int c = (tid & 31) * 2;
        float2 v = *(const float2*)(x + ((size_t)(wg * BT + r) * kT + t) * kIN + c);
        xs[r * kIN + c]     = (_Float16)v.x;
        xs[r * kIN + c + 1] = (_Float16)v.y;
      }
      __syncthreads();
      // u GEMM
      {
        v8f au = vbroadcast(biU);
#pragma unroll
        for (int k = 0; k < 2; ++k) {
          v16h A = load_frag_a(xs + mRow * kIN, k * 32);
          au = wmma16(A, load_frag_b(wWin + (size_t)(w * 16 + nIdx) * kIN, k * 32), au);
        }
#pragma unroll
        for (int v = 0; v < 8; ++v)
          us[(v + mBase) * kH + w * 16 + nIdx] = (_Float16)au[v];
      }
      __syncthreads();
      // ig GEMM
      {
        v8f aR = vbroadcast(biR), aZ = vbroadcast(biZ), aN = vbroadcast(biN);
        const _Float16* rR = wWih + (size_t)((0 * 16 + w) * 16 + nIdx) * kH;
        const _Float16* rZ = wWih + (size_t)((16 + w) * 16 + nIdx) * kH;
        const _Float16* rN = wWih + (size_t)((32 + w) * 16 + nIdx) * kH;
#pragma unroll
        for (int k = 0; k < 8; ++k) {
          v16h A = load_frag_a(us + mRow * kH, k * 32);
          aR = wmma16(A, load_frag_b(rR, k * 32), aR);
          aZ = wmma16(A, load_frag_b(rZ, k * 32), aZ);
          aN = wmma16(A, load_frag_b(rN, k * 32), aN);
        }
#pragma unroll
        for (int v = 0; v < 8; ++v) { igr[v] = aR[v]; igz[v] = aZ[v]; ign[v] = aN[v]; }
      }
    }

    // recurrent GEMM: A fragments from current h buffer, B pinned in VGPRs
#pragma unroll
    for (int k = 0; k < 8; ++k) {
      v16h A = load_frag_a(hcur + mRow * kH, k * 32);
      accR = wmma16(A, Bhh[0][k], accR);
      accZ = wmma16(A, Bhh[1][k], accZ);
      accN = wmma16(A, Bhh[2][k], accN);
    }

    // GRU gate math + h update (wave w owns hidden cols w*16..w*16+15)
#pragma unroll
    for (int v = 0; v < 8; ++v) {
      float ir = USE_IG ? (float)r8c[v] : igr[v];
      float iz = USE_IG ? (float)z8c[v] : igz[v];
      float in_ = USE_IG ? (float)n8c[v] : ign[v];
      float r = sigm(ir + accR[v]);
      float z = sigm(iz + accZ[v]);
      float n = tanh_f(in_ + r * (accN[v] + bn));
      float hn = n + z * (hold[v] - n);
      hold[v] = hn;
      hnxt[(v + mBase) * kH + w * 16 + nIdx] = (_Float16)hn;
    }
    __syncthreads();   // new h visible; also protects buffer reuse 2 steps later
  }

  // output: out = h_final @ W_out^T + b_out (waves 0,1 -> 2 N-tiles of 16)
  const _Float16* hf = hs + (kT & 1) * (BT * kH);
  if (w < 2) {
    v8f acc = vbroadcast(b_out[w * 16 + nIdx]);
    const _Float16* row = wWout + (size_t)(w * 16 + nIdx) * kH;
#pragma unroll
    for (int k = 0; k < 8; ++k) {
      v16h A = load_frag_a(hf + mRow * kH, k * 32);
      acc = wmma16(A, load_frag_b(row, k * 32), acc);
    }
#pragma unroll
    for (int v = 0; v < 8; ++v)
      out[(size_t)(wg * BT + v + mBase) * kOUT + w * 16 + nIdx] = acc[v];
  }
}

} // anonymous namespace

// ---------------------------------------------------------------------------
extern "C" void kernel_launch(void* const* d_in, const int* in_sizes, int n_in,
                              void* d_out, int out_size, void* d_ws, size_t ws_size,
                              hipStream_t stream) {
  const float* x     = (const float*)d_in[0];
  const float* W_in  = (const float*)d_in[1];
  const float* b_in  = (const float*)d_in[2];
  const float* W_ih  = (const float*)d_in[3];
  const float* W_hh  = (const float*)d_in[4];
  const float* b_ih  = (const float*)d_in[5];
  const float* b_n   = (const float*)d_in[6];
  const float* W_out = (const float*)d_in[7];
  const float* b_out = (const float*)d_in[8];
  float* out = (float*)d_out;

  char* ws = (char*)d_ws;
  _Float16* wWin  = (_Float16*)(ws + WS_WIN);
  _Float16* wWih  = (_Float16*)(ws + WS_WIH);
  _Float16* wWhh  = (_Float16*)(ws + WS_WHH);
  _Float16* wWout = (_Float16*)(ws + WS_WOUT);
  h8*       igws  = (h8*)(ws + WS_IG);

  // phase0: weight conversion (covers largest array kG3*kH)
  cvt_weights_kernel<<<(kG3 * kH + 255) / 256, 256, 0, stream>>>(
      W_in, W_ih, W_hh, W_out, wWin, wWih, wWhh, wWout);

  const bool big = ws_size >= WS_IG + IG_BYTES;
  if (big) {
    ig_kernel<<<kT * NWG, THREADS, 0, stream>>>(x, b_in, b_ih, wWin, wWih, igws);
    rec_kernel<true><<<NWG, THREADS, 0, stream>>>(
        x, b_in, b_ih, b_n, b_out, wWin, wWih, wWhh, wWout, igws, out);
  } else {
    rec_kernel<false><<<NWG, THREADS, 0, stream>>>(
        x, b_in, b_ih, b_n, b_out, wWin, wWih, wWhh, wWout, igws, out);
  }
}
